// MultiQueryAttention_72602127172297
// MI455X (gfx1250) — matrix operand order repaired and checked by reading, not compiled, not run on hardware
//
#include <hip/hip_runtime.h>
#include <hip/hip_bf16.h>

// ---------------------------------------------------------------------------
// MQA for MI455X (gfx1250): bf16 WMMA 16x16x32 with f32 accumulation.
// Phase 1: project x -> Q[B,H,T,64], K[B,T,64], V^T[B,64,T] (all bf16).
// Phase 2: flash-attention with online softmax, WMMA for S=QK^T and O=PV.
// ---------------------------------------------------------------------------

typedef __attribute__((ext_vector_type(16))) __bf16 v16bf;
typedef __attribute__((ext_vector_type(8)))  __bf16 v8bf;
typedef __attribute__((ext_vector_type(8)))  float  v8f;

#define B_DIM 4
#define T_DIM 2048
#define D_DIM 1024
#define H_DIM 16
#define HD    64

// =====================  Phase 1: QKV projection  ===========================
// grid.x = (B*T)/128 row blocks, grid.y = 18 units (0=K, 1=V, 2+h=Q head h).
// Each wave: 16 rows x 64 cols, K-loop over D in chunks of 32.
__global__ __launch_bounds__(256) void qkv_proj_kernel(
    const float* __restrict__ x,   // [B*T, D]
    const float* __restrict__ Wq,  // [H, D, HD]
    const float* __restrict__ bq,  // [H, HD]
    const float* __restrict__ Wk,  // [D, HD]
    const float* __restrict__ bk,  // [HD]
    const float* __restrict__ Wv,  // [D, HD]
    const float* __restrict__ bv,  // [HD]
    __bf16* __restrict__ Qb,       // [B*H*T, HD]
    __bf16* __restrict__ Kb,       // [B*T, HD]
    __bf16* __restrict__ Vt)       // [B, HD, T]
{
    __shared__ __align__(64) __bf16 wt[64][32];   // W^T chunk: [n][k], 4 KB

    const int unit = blockIdx.y;
    const int rowStart = blockIdx.x * 128;
    const int tid  = threadIdx.x;
    const int wave = tid >> 5;
    const int lane = tid & 31;
    const int half = lane >> 4;
    const int m    = lane & 15;
    const int rbase = rowStart + wave * 16;

    const float* W;
    const float* bias;
    if (unit == 0)      { W = Wk; bias = bk; }
    else if (unit == 1) { W = Wv; bias = bv; }
    else                { W = Wq + (size_t)(unit - 2) * D_DIM * HD;
                          bias = bq + (unit - 2) * HD; }

    v8f acc[4] = {};

    const int wn = tid >> 2;        // 0..63 : output col
    const int wk = (tid & 3) * 8;   // 0,8,16,24 : k offset

    for (int k0 = 0; k0 < D_DIM; k0 += 32) {
        // stage W^T chunk to LDS as bf16 (wt[n][k] = W[k0+k][n])
        #pragma unroll
        for (int i = 0; i < 8; ++i)
            wt[wn][wk + i] = (__bf16)W[(size_t)(k0 + wk + i) * HD + wn];
        __syncthreads();

        // A operand: 16 rows x 32 K of x, bf16.
        // lanes 0-15: K {0..15}; lanes 16-31: K {16..31}
        const float* xrow = x + (size_t)(rbase + m) * D_DIM + k0;
        v16bf a;
        #pragma unroll
        for (int j = 0; j < 8; ++j) a[j]     = (__bf16)xrow[half * 16 + j];
        #pragma unroll
        for (int j = 0; j < 8; ++j) a[8 + j] = (__bf16)xrow[half * 16 + 8 + j];

        #pragma unroll
        for (int nt = 0; nt < 4; ++nt) {
            // B operand: col = nt*16+m, element j <-> K = half*16+j
            v16bf bvv = *(const v16bf*)&wt[nt * 16 + m][half * 16];
            acc[nt] = __builtin_amdgcn_wmma_f32_16x16x32_bf16(
                false, a, false, bvv, (short)0, acc[nt], false, false);
        }
        __syncthreads();
    }

    // Epilogue: add bias, store bf16 to unit-specific layout.
    // D layout: element r of lane -> row M = half*8+r, col N = nt*16 + m
    #pragma unroll
    for (int nt = 0; nt < 4; ++nt) {
        const int col = nt * 16 + m;
        const float bia = bias[col];
        #pragma unroll
        for (int r = 0; r < 8; ++r) {
            const int row = rbase + half * 8 + r;     // global b*T + t
            const __bf16 v = (__bf16)(acc[nt][r] + bia);
            if (unit == 0) {
                Kb[(size_t)row * HD + col] = v;
            } else if (unit == 1) {
                const int b = row >> 11;
                const int t = row & (T_DIM - 1);
                Vt[((size_t)b * HD + col) * T_DIM + t] = v;
            } else {
                const int b = row >> 11;
                const int t = row & (T_DIM - 1);
                const int h = unit - 2;
                Qb[(((size_t)(b * H_DIM + h) * T_DIM) + t) * HD + col] = v;
            }
        }
    }
}

// =====================  Phase 2: flash attention  ==========================
// grid.x = T/128 query blocks, grid.y = B*H.  8 waves, 16 q-rows per wave.
__global__ __launch_bounds__(256) void mqa_attn_kernel(
    const __bf16* __restrict__ Qb,  // [B*H*T, 64]
    const __bf16* __restrict__ Kb,  // [B*T, 64]
    const __bf16* __restrict__ Vt,  // [B, 64, T]
    float* __restrict__ out)        // [B, T, H*64]
{
    __shared__ __align__(64) __bf16 Kl[64][64];      // [key][hd], 8 KB
    __shared__ __align__(64) __bf16 Vl[64][64];      // [e][key],  8 KB
    __shared__ __align__(64) __bf16 Pl[8][16][64];   // per-wave [m][key], 16 KB

    const int bh   = blockIdx.y;
    const int b    = bh >> 4;
    const int h    = bh & 15;
    const int tid  = threadIdx.x;
    const int wave = tid >> 5;
    const int lane = tid & 31;
    const int half = lane >> 4;
    const int m    = lane & 15;

    const int q0 = blockIdx.x * 128 + wave * 16;     // wave's query base in T

    // Preload this wave's Q tile (16 x 64) as two A-operand chunks.
    const __bf16* qrow = Qb + ((size_t)bh * T_DIM + q0 + m) * HD;
    v16bf qa[2];
    #pragma unroll
    for (int kc = 0; kc < 2; ++kc) {
        v8bf lo = *(const v8bf*)(qrow + kc * 32 + half * 16);
        v8bf hi = *(const v8bf*)(qrow + kc * 32 + half * 16 + 8);
        #pragma unroll
        for (int j = 0; j < 8; ++j) { qa[kc][j] = lo[j]; qa[kc][8 + j] = hi[j]; }
    }

    const float sc = 0.125f * 1.44269504088896340736f;  // 1/sqrt(64) * log2(e)

    float mrow[8], lrow[8];
    v8f acc_o[4] = {};
    #pragma unroll
    for (int r = 0; r < 8; ++r) { mrow[r] = -1.0e30f; lrow[r] = 0.0f; }

    // cooperative-staging indices: each thread copies 16 contiguous bf16
    const int sIdx = tid * 16;
    const int sRow = sIdx >> 6;       // 0..63
    const int sOff = sIdx & 63;       // 0,16,32,48

    for (int kbase = 0; kbase < T_DIM; kbase += 64) {
        // Stage K tile [key][hd] and V^T tile [e][key] (both contiguous 32B).
        *(v16bf*)&Kl[sRow][sOff] =
            *(const v16bf*)(Kb + ((size_t)(b * T_DIM + kbase + sRow)) * HD + sOff);
        *(v16bf*)&Vl[sRow][sOff] =
            *(const v16bf*)(Vt + ((size_t)b * HD + sRow) * T_DIM + kbase + sOff);
        __syncthreads();

        if (kbase + 64 < T_DIM) {   // prefetch next tiles -> global_prefetch_b8
            __builtin_prefetch(Kb + ((size_t)(b * T_DIM + kbase + 64 + sRow)) * HD + sOff, 0, 1);
            __builtin_prefetch(Vt + ((size_t)b * HD + sRow) * T_DIM + kbase + 64 + sOff, 0, 1);
        }

        // ---- S = Q * K^T  (16 x 64), 8 WMMAs ----
        v8f s[4] = {};
        #pragma unroll
        for (int nt = 0; nt < 4; ++nt) {
            #pragma unroll
            for (int kc = 0; kc < 2; ++kc) {
                v16bf kb = *(const v16bf*)&Kl[nt * 16 + m][kc * 32 + half * 16];
                s[nt] = __builtin_amdgcn_wmma_f32_16x16x32_bf16(
                    false, qa[kc], false, kb, (short)0, s[nt], false, false);
            }
        }
        // scale into log2 domain
        #pragma unroll
        for (int nt = 0; nt < 4; ++nt)
            #pragma unroll
            for (int r = 0; r < 8; ++r) s[nt][r] *= sc;

        // ---- online softmax (row = half*8+r spans the 16 lanes of a half) ----
        #pragma unroll
        for (int r = 0; r < 8; ++r) {
            float mx = s[0][r];
            mx = fmaxf(mx, s[1][r]); mx = fmaxf(mx, s[2][r]); mx = fmaxf(mx, s[3][r]);
            #pragma unroll
            for (int off = 8; off >= 1; off >>= 1)
                mx = fmaxf(mx, __shfl_xor(mx, off, 16));
            const float mnew = fmaxf(mrow[r], mx);
            const float corr = exp2f(mrow[r] - mnew);
            mrow[r] = mnew;
            float rowsum = 0.0f;
            #pragma unroll
            for (int nt = 0; nt < 4; ++nt) {
                const float p = exp2f(s[nt][r] - mnew);
                s[nt][r] = p;
                rowsum += p;
            }
            #pragma unroll
            for (int off = 8; off >= 1; off >>= 1)
                rowsum += __shfl_xor(rowsum, off, 16);
            lrow[r] = lrow[r] * corr + rowsum;
            acc_o[0][r] *= corr; acc_o[1][r] *= corr;
            acc_o[2][r] *= corr; acc_o[3][r] *= corr;
        }

        // ---- D-layout -> A-layout for P via per-wave LDS ----
        #pragma unroll
        for (int nt = 0; nt < 4; ++nt)
            #pragma unroll
            for (int r = 0; r < 8; ++r)
                Pl[wave][half * 8 + r][nt * 16 + m] = (__bf16)s[nt][r];
        // (same-wave LDS ops are in-order; no cross-wave sharing of Pl)

        // ---- O += P * V  (16 x 64), 8 WMMAs ----
        #pragma unroll
        for (int kc = 0; kc < 2; ++kc) {
            const __bf16* pp = &Pl[wave][m][kc * 32];
            v8bf lo = *(const v8bf*)(pp + half * 16);
            v8bf hi = *(const v8bf*)(pp + half * 16 + 8);
            v16bf pa;
            #pragma unroll
            for (int j = 0; j < 8; ++j) { pa[j] = lo[j]; pa[8 + j] = hi[j]; }
            #pragma unroll
            for (int et = 0; et < 4; ++et) {
                v16bf vb = *(const v16bf*)&Vl[et * 16 + m][kc * 32 + half * 16];
                acc_o[et] = __builtin_amdgcn_wmma_f32_16x16x32_bf16(
                    false, pa, false, vb, (short)0, acc_o[et], false, false);
            }
        }
        __syncthreads();
    }

    // ---- finalize: O / l, scatter fp32 to [B, T, H*64] ----
    #pragma unroll
    for (int et = 0; et < 4; ++et) {
        #pragma unroll
        for (int r = 0; r < 8; ++r) {
            const int q = q0 + half * 8 + r;
            const float o = acc_o[et][r] / lrow[r];
            out[((size_t)b * T_DIM + q) * (H_DIM * HD) + h * HD + et * 16 + m] = o;
        }
    }
}

// ===========================  launch  ======================================
extern "C" void kernel_launch(void* const* d_in, const int* in_sizes, int n_in,
                              void* d_out, int out_size, void* d_ws, size_t ws_size,
                              hipStream_t stream) {
    (void)in_sizes; (void)n_in; (void)out_size; (void)ws_size;
    const float* x  = (const float*)d_in[0];
    const float* Wq = (const float*)d_in[1];
    const float* bq = (const float*)d_in[2];
    const float* Wk = (const float*)d_in[3];
    const float* bk = (const float*)d_in[4];
    const float* Wv = (const float*)d_in[5];
    const float* bv = (const float*)d_in[6];
    float* out = (float*)d_out;

    char* ws = (char*)d_ws;
    const size_t qBytes = (size_t)B_DIM * H_DIM * T_DIM * HD * sizeof(__bf16); // 16 MB
    const size_t kBytes = (size_t)B_DIM * T_DIM * HD * sizeof(__bf16);         // 1 MB
    __bf16* Qb = (__bf16*)ws;
    __bf16* Kb = (__bf16*)(ws + qBytes);
    __bf16* Vt = (__bf16*)(ws + qBytes + kBytes);

    dim3 blk(256);
    dim3 g1((B_DIM * T_DIM) / 128, 18);   // 64 row-blocks x 18 units
    qkv_proj_kernel<<<g1, blk, 0, stream>>>(x, Wq, bq, Wk, bk, Wv, bv, Qb, Kb, Vt);

    dim3 g2(T_DIM / 128, B_DIM * H_DIM);  // 16 q-blocks x 64 (b,h)
    mqa_attn_kernel<<<g2, blk, 0, stream>>>(Qb, Kb, Vt, out);
}
